// DecisionTransformer_70798240907787
// MI455X (gfx1250) — compile-verified, matrix operand order checked by reading, not run on hardware
//
#include <hip/hip_runtime.h>
#include <hip/hip_bf16.h>

// ---------------- constants ----------------
#define BB 128          // batch
#define TT 64           // timesteps
#define SS 192          // sequence = 3T
#define HH 512          // hidden
#define NHD 8           // heads
#define HDD 64          // head dim
#define MM (BB*SS)      // 24576 GEMM rows

typedef __attribute__((ext_vector_type(16))) __bf16 v16bf;
typedef __attribute__((ext_vector_type(8)))  float  v8f;

// Load a WMMA 16-bit A/B operand fragment from a row-major matrix.
// Lane r (0-15) holds row/col r; K pairs (k0,k0+1) are contiguous in memory.
// half = lane>>4 selects the K sub-range per the CDNA5 16-bit A layout.
__device__ __forceinline__ v16bf afrag(const __bf16* rowptr, int kbase, int half) {
  v16bf a;
#pragma unroll
  for (int v = 0; v < 8; ++v) {
    int k0 = kbase + ((v & 4) ? 16 : 0) + half * 8 + 2 * (v & 3);
    a[2 * v]     = rowptr[k0];
    a[2 * v + 1] = rowptr[k0 + 1];
  }
  return a;
}

__device__ __forceinline__ v8f wmma_bf16(v16bf a, v16bf b, v8f c) {
  return __builtin_amdgcn_wmma_f32_16x16x32_bf16(false, a, false, b, (short)0, c,
                                                 false, false);
}

// ---------------- weight transpose + fp32->bf16 convert (once per launch) ----------------
// W fp32 [K=512][N=512] row-major  ->  Wt bf16 [N=512][K=512] row-major.
__global__ __launch_bounds__(256)
void wconv_kernel(const float* __restrict__ W, __bf16* __restrict__ Wt) {
  int g = blockIdx.x * 256 + threadIdx.x;   // grid 1024 -> 262144 elements
  int k = g >> 9, n = g & 511;
  Wt[(size_t)n * HH + k] = (__bf16)W[g];
}

// ---------------- embedding + initial LayerNorm ----------------
// grid = B*T blocks, 256 threads. Builds x[b, 3t+{0,1,2}, :] (fp32 residual stream).
__global__ __launch_bounds__(256)
void embed_ln_kernel(const float* __restrict__ states, const float* __restrict__ actions,
                     const float* __restrict__ rtg, const int* __restrict__ timesteps,
                     const float* __restrict__ Wr, const float* __restrict__ br,
                     const float* __restrict__ Ws, const float* __restrict__ bs,
                     const float* __restrict__ Wa, const float* __restrict__ ba,
                     const float* __restrict__ te_tab,
                     const float* __restrict__ g, const float* __restrict__ be,
                     float* __restrict__ x) {
  int bt = blockIdx.x;
  int b = bt / TT, t = bt % TT;
  int ts = timesteps[bt];
  const float* st = states + (size_t)bt * 17;
  const float* ac = actions + (size_t)bt * 6;
  float rv = rtg[bt];
  float vals[3][2];
#pragma unroll
  for (int u = 0; u < 2; ++u) {
    int h = threadIdx.x + u * 256;
    float te = te_tab[(size_t)ts * HH + h];
    vals[0][u] = rv * Wr[h] + br[h] + te;              // return token
    float s = bs[h];
#pragma unroll
    for (int i = 0; i < 17; ++i) s += st[i] * Ws[i * HH + h];
    vals[1][u] = s + te;                               // state token
    float aa = ba[h];
#pragma unroll
    for (int i = 0; i < 6; ++i) aa += ac[i] * Wa[i * HH + h];
    vals[2][u] = aa + te;                              // action token
  }
  __shared__ float red[256];
  for (int tok = 0; tok < 3; ++tok) {
    red[threadIdx.x] = vals[tok][0] + vals[tok][1];
    __syncthreads();
    for (int s2 = 128; s2 > 0; s2 >>= 1) {
      if (threadIdx.x < s2) red[threadIdx.x] += red[threadIdx.x + s2];
      __syncthreads();
    }
    float mean = red[0] * (1.f / HH);
    __syncthreads();
    float d0 = vals[tok][0] - mean, d1 = vals[tok][1] - mean;
    red[threadIdx.x] = d0 * d0 + d1 * d1;
    __syncthreads();
    for (int s2 = 128; s2 > 0; s2 >>= 1) {
      if (threadIdx.x < s2) red[threadIdx.x] += red[threadIdx.x + s2];
      __syncthreads();
    }
    float rstd = rsqrtf(red[0] * (1.f / HH) + 1e-5f);
    __syncthreads();
    size_t row = (size_t)(b * SS + 3 * t + tok) * HH;
#pragma unroll
    for (int u = 0; u < 2; ++u) {
      int h = threadIdx.x + u * 256;
      x[row + h] = (vals[tok][u] - mean) * rstd * g[h] + be[h];
    }
  }
}

// ---------------- LayerNorm: fp32 in -> bf16 out (GEMM A operand) ----------------
// grid = M/8, 256 threads (8 waves, one row per wave).
__global__ __launch_bounds__(256)
void ln_bf16_kernel(const float* __restrict__ x, const float* __restrict__ g,
                    const float* __restrict__ b, __bf16* __restrict__ out) {
  int wid = threadIdx.x >> 5, lane = threadIdx.x & 31;
  int row = blockIdx.x * 8 + wid;
  const float* xr = x + (size_t)row * HH;
  float v[16];
  float s = 0.f;
#pragma unroll
  for (int j = 0; j < 16; ++j) { v[j] = xr[lane + 32 * j]; s += v[j]; }
#pragma unroll
  for (int off = 16; off >= 1; off >>= 1) s += __shfl_xor(s, off, 32);
  float mean = s * (1.f / HH);
  float q = 0.f;
#pragma unroll
  for (int j = 0; j < 16; ++j) { float d = v[j] - mean; q += d * d; }
#pragma unroll
  for (int off = 16; off >= 1; off >>= 1) q += __shfl_xor(q, off, 32);
  float rstd = rsqrtf(q * (1.f / HH) + 1e-5f);
  __bf16* orow = out + (size_t)row * HH;
#pragma unroll
  for (int j = 0; j < 16; ++j) {
    int c = lane + 32 * j;
    orow[c] = (__bf16)((v[j] - mean) * rstd * g[c] + b[c]);
  }
}

// ---------------- bf16 WMMA GEMM: [M x 512] x [512 x 512] ----------------
// A bf16 [M][512], Wt bf16 [N][K] (pre-transposed). Block tile 128x128, 8 waves,
// wave tile 32x64 (2x4 WMMA tiles, A-frags reused across 4 B-frags).
// MODE 0: +bias, scatter bf16 to [b,h,s,d] (QKV)
// MODE 1: +bias, exact GELU, bf16 [M x 512]
// MODE 2: +bias, += residual, fp32 in place
template <int MODE>
__global__ __launch_bounds__(256)
void gemm512_kernel(const __bf16* __restrict__ A, const __bf16* __restrict__ Wt,
                    const float* __restrict__ bias, float* __restrict__ resid,
                    __bf16* __restrict__ outb) {
  __shared__ alignas(16) __bf16 As[128][32];
  __shared__ alignas(16) __bf16 Bs[128][32];  // [n][k]
  int tid = threadIdx.x;
  int lane = tid & 31, half = lane >> 4, r = lane & 15;
  int wid = tid >> 5, wm = wid & 3, wn = wid >> 2;
  int bm = blockIdx.x * 128, bn = blockIdx.y * 128;
  int rowT = tid >> 1, colT = (tid & 1) * 16;   // staging assignment
  const __bf16* gA = A  + (size_t)(bm + rowT) * HH + colT;
  const __bf16* gB = Wt + (size_t)(bn + rowT) * HH + colT;

  v8f c[2][4];
#pragma unroll
  for (int i = 0; i < 2; ++i)
#pragma unroll
    for (int j = 0; j < 4; ++j)
#pragma unroll
      for (int vv = 0; vv < 8; ++vv) c[i][j][vv] = 0.f;

  for (int kb = 0; kb < HH; kb += 32) {
    { // stage A tile and B tile (both bf16 row-major, 32B per thread each)
      const uint4* sa = (const uint4*)(gA + kb);
      uint4* da = (uint4*)(&As[rowT][colT]);
      da[0] = sa[0]; da[1] = sa[1];
      const uint4* sb = (const uint4*)(gB + kb);
      uint4* db = (uint4*)(&Bs[rowT][colT]);
      db[0] = sb[0]; db[1] = sb[1];
      if (kb + 32 < HH) {   // prefetch next K tile (global_prefetch_b8)
        __builtin_prefetch(gA + kb + 32, 0, 3);
        __builtin_prefetch(gB + kb + 32, 0, 3);
      }
    }
    __syncthreads();
    v16bf af[2], bf[4];
#pragma unroll
    for (int i = 0; i < 2; ++i) af[i] = afrag(&As[wm * 32 + i * 16 + r][0], 0, half);
#pragma unroll
    for (int j = 0; j < 4; ++j) bf[j] = afrag(&Bs[wn * 64 + j * 16 + r][0], 0, half);
#pragma unroll
    for (int i = 0; i < 2; ++i)
#pragma unroll
      for (int j = 0; j < 4; ++j) c[i][j] = wmma_bf16(af[i], bf[j], c[i][j]);
    __syncthreads();
  }
  // epilogue
#pragma unroll
  for (int i = 0; i < 2; ++i)
#pragma unroll
    for (int j = 0; j < 4; ++j) {
      int colt = bn + wn * 64 + j * 16 + r;
      float bv = bias[colt];
#pragma unroll
      for (int vv = 0; vv < 8; ++vv) {
        int row = bm + wm * 32 + i * 16 + vv + half * 8;
        float val = c[i][j][vv] + bv;
        if (MODE == 0) {
          int bq = row / SS, s = row % SS, hh = colt >> 6, d = colt & 63;
          outb[(((size_t)(bq * NHD + hh) * SS + s) << 6) + d] = (__bf16)val;
        } else if (MODE == 1) {
          val = 0.5f * val * (1.f + erff(val * 0.70710678118f));
          outb[(size_t)row * HH + colt] = (__bf16)val;
        } else {
          float* p = resid + (size_t)row * HH + colt;
          *p = *p + val;
        }
      }
    }
}

// ---------------- attention: one block per (b, head), 6 waves x 32 query rows ----------------
__global__ __launch_bounds__(192)
void attn_kernel(const __bf16* __restrict__ q, const __bf16* __restrict__ k,
                 const __bf16* __restrict__ v, const int* __restrict__ mask,
                 __bf16* __restrict__ y) {
  __shared__ alignas(16) __bf16 Vt[HDD * SS];     // V transposed: [d][s]
  __shared__ alignas(16) __bf16 Pb[12 * 16 * 32]; // per-wave P scratch (6w x 2i x 16x32)
  __shared__ float addm[SS];
  int bh = blockIdx.x, b = bh >> 3, hh = bh & 7;
  const __bf16* qg = q + (size_t)bh * SS * HDD;
  const __bf16* kg = k + (size_t)bh * SS * HDD;
  const __bf16* vg = v + (size_t)bh * SS * HDD;
  int tid = threadIdx.x;
  for (int idx = tid; idx < SS * HDD; idx += 192) {
    int s = idx >> 6, d = idx & 63;
    Vt[d * SS + s] = vg[idx];
  }
  addm[tid] = (1.f - (float)mask[b * TT + tid / 3]) * -10000.f;
  __syncthreads();

  int w = tid >> 5, lane = tid & 31, half = lane >> 4, r = lane & 15;
  int qbase = w * 32;
  v16bf qf[2][2];
#pragma unroll
  for (int i = 0; i < 2; ++i)
#pragma unroll
    for (int dh = 0; dh < 2; ++dh)
      qf[i][dh] = afrag(qg + (size_t)(qbase + i * 16 + r) * HDD, dh * 32, half);

  float mrun[2][8], lrun[2][8];
  v8f O[2][4];
#pragma unroll
  for (int i = 0; i < 2; ++i)
#pragma unroll
    for (int vv = 0; vv < 8; ++vv) { mrun[i][vv] = -1e30f; lrun[i][vv] = 0.f; }
#pragma unroll
  for (int i = 0; i < 2; ++i)
#pragma unroll
    for (int nt = 0; nt < 4; ++nt)
#pragma unroll
      for (int vv = 0; vv < 8; ++vv) O[i][nt][vv] = 0.f;

  for (int ks = 0; ks <= w; ++ks) {   // causal: wave w only needs keys <= qbase+31
    int kb = ks * 32;
    v16bf kf[2][2];
#pragma unroll
    for (int jj = 0; jj < 2; ++jj)
#pragma unroll
      for (int dh = 0; dh < 2; ++dh)
        kf[jj][dh] = afrag(kg + (size_t)(kb + jj * 16 + r) * HDD, dh * 32, half);

#pragma unroll
    for (int i = 0; i < 2; ++i) {
      v8f s0 = {0.f, 0.f, 0.f, 0.f, 0.f, 0.f, 0.f, 0.f};
      v8f s1 = {0.f, 0.f, 0.f, 0.f, 0.f, 0.f, 0.f, 0.f};
#pragma unroll
      for (int dh = 0; dh < 2; ++dh) {
        s0 = wmma_bf16(qf[i][dh], kf[0][dh], s0);
        s1 = wmma_bf16(qf[i][dh], kf[1][dh], s1);
      }
#pragma unroll
      for (int vv = 0; vv < 8; ++vv) {
        int row = qbase + i * 16 + vv + half * 8;
        int c0 = kb + r, c1 = c0 + 16;
        float x0 = s0[vv] * 0.125f;   // 1/sqrt(64)
        float x1 = s1[vv] * 0.125f;
        x0 = (c0 <= row ? x0 : -10000.f) + addm[c0];
        x1 = (c1 <= row ? x1 : -10000.f) + addm[c1];
        float t = fmaxf(x0, x1);
#pragma unroll
        for (int off = 8; off >= 1; off >>= 1) t = fmaxf(t, __shfl_xor(t, off, 32));
        float mo = mrun[i][vv];
        float mn = fmaxf(mo, t);
        float corr = __expf(mo - mn);
        mrun[i][vv] = mn;
        float p0 = __expf(x0 - mn), p1 = __expf(x1 - mn);
        float rs = p0 + p1;
#pragma unroll
        for (int off = 8; off >= 1; off >>= 1) rs += __shfl_xor(rs, off, 32);
        lrun[i][vv] = lrun[i][vv] * corr + rs;
#pragma unroll
        for (int nt = 0; nt < 4; ++nt) O[i][nt][vv] *= corr;
        int pb = ((w * 2 + i) * 16 + vv + half * 8) * 32;
        Pb[pb + r] = (__bf16)p0;
        Pb[pb + 16 + r] = (__bf16)p1;
      }
    }
    // P @ V (wave-private LDS scratch; LDS ops are in-order within a wave)
    v16bf vf[4];
#pragma unroll
    for (int nt = 0; nt < 4; ++nt) vf[nt] = afrag(&Vt[(nt * 16 + r) * SS], kb, half);
#pragma unroll
    for (int i = 0; i < 2; ++i) {
      v16bf pf = afrag(&Pb[((w * 2 + i) * 16 + r) * 32], 0, half);
#pragma unroll
      for (int nt = 0; nt < 4; ++nt) O[i][nt] = wmma_bf16(pf, vf[nt], O[i][nt]);
    }
  }
  // normalize and emit bf16 y in [M x 512] layout (feeds Wo GEMM)
#pragma unroll
  for (int i = 0; i < 2; ++i)
#pragma unroll
    for (int vv = 0; vv < 8; ++vv) {
      int row = qbase + i * 16 + vv + half * 8;
      float rl = 1.f / lrun[i][vv];
#pragma unroll
      for (int nt = 0; nt < 4; ++nt) {
        int d = nt * 16 + r;
        y[((size_t)(b * SS + row)) * HH + hh * HDD + d] = (__bf16)(O[i][nt][vv] * rl);
      }
    }
}

// ---------------- prediction heads (N=17 and N=6, VALU) ----------------
__global__ __launch_bounds__(32)
void heads_kernel(const float* __restrict__ x, const float* __restrict__ Wps,
                  const float* __restrict__ bps, const float* __restrict__ Wpa,
                  const float* __restrict__ bpa, float* __restrict__ out) {
  int bt = blockIdx.x;
  int b = bt >> 6, t = bt & 63;
  int j = threadIdx.x;
  const float* xs = x + (size_t)(b * SS + 3 * t + 2) * HH; // action token -> state preds
  const float* xa = x + (size_t)(b * SS + 3 * t + 1) * HH; // state token -> action preds
  if (j < 17) {
    float s = bps[j];
    for (int kk = 0; kk < HH; ++kk) s += xs[kk] * Wps[kk * 17 + j];
    out[(size_t)bt * 17 + j] = s;
  } else if (j < 23) {
    int jj = j - 17;
    float s = bpa[jj];
    for (int kk = 0; kk < HH; ++kk) s += xa[kk] * Wpa[kk * 6 + jj];
    out[(size_t)(BB * TT * 17) + (size_t)bt * 6 + jj] = s;
  }
}

// ---------------- launch ----------------
// params leaf order (insertion order of setup_inputs dict):
// 0 states 1 actions 2 rewards 3 returns_to_go 4 timesteps 5 attention_mask
// 6 embed_timestep 7 W_return 8 b_return 9 W_reward 10 b_reward 11 W_state 12 b_state
// 13 W_action 14 b_action 15 ln_g 16 ln_b
// blocks: 17 + 16*L + {0 ln1_g,1 ln1_b,2 Wq,3 bq,4 Wk,5 bk,6 Wv,7 bv,8 Wo,9 bo,
//                      10 ln2_g,11 ln2_b,12 W1,13 b1,14 W2,15 b2}
// 113 W_ps 114 b_ps 115 W_pa 116 b_pa
extern "C" void kernel_launch(void* const* d_in, const int* in_sizes, int n_in,
                              void* d_out, int out_size, void* d_ws, size_t ws_size,
                              hipStream_t stream) {
  (void)in_sizes; (void)n_in; (void)out_size; (void)ws_size;
  const float* F[128];
  for (int i = 0; i < 117 && i < n_in; ++i) F[i] = (const float*)d_in[i];
  const int* timesteps = (const int*)d_in[4];
  const int* attn_mask = (const int*)d_in[5];

  char* ws = (char*)d_ws;
  float*  x  = (float*)ws;                                 // residual stream, fp32
  __bf16* a  = (__bf16*)(ws + (size_t)50331648);           // LN1 out; aliased as attn-out y
  __bf16* qb = (__bf16*)(ws + (size_t)75497472);           // Q; aliased as LN2 out
  __bf16* kb = (__bf16*)(ws + (size_t)100663296);          // K; aliased as GELU out
  __bf16* vb = (__bf16*)(ws + (size_t)125829120);          // V
  __bf16* wt = (__bf16*)(ws + (size_t)150994944);          // 36 transposed bf16 weights

  // pre-transpose/convert all 6x6 weight matrices (bf16, [N][K]) -- L2 resident
  static const int widx[6] = {2, 4, 6, 8, 12, 14};
  for (int L = 0; L < 6; ++L)
    for (int m = 0; m < 6; ++m)
      wconv_kernel<<<1024, 256, 0, stream>>>(F[17 + 16 * L + widx[m]],
                                             wt + (size_t)(L * 6 + m) * HH * HH);

  embed_ln_kernel<<<BB * TT, 256, 0, stream>>>(
      F[0], F[1], F[3], timesteps, F[7], F[8], F[11], F[12], F[13], F[14],
      F[6], F[15], F[16], x);

  dim3 ggrid(MM / 128, HH / 128);
  for (int L = 0; L < 6; ++L) {
    const float* const* Bp = &F[17 + 16 * L];
    const __bf16* Wl = wt + (size_t)L * 6 * HH * HH;
    ln_bf16_kernel<<<MM / 8, 256, 0, stream>>>(x, Bp[0], Bp[1], a);
    gemm512_kernel<0><<<ggrid, 256, 0, stream>>>(a, Wl + 0 * HH * HH, Bp[3], nullptr, qb);
    gemm512_kernel<0><<<ggrid, 256, 0, stream>>>(a, Wl + 1 * HH * HH, Bp[5], nullptr, kb);
    gemm512_kernel<0><<<ggrid, 256, 0, stream>>>(a, Wl + 2 * HH * HH, Bp[7], nullptr, vb);
    attn_kernel<<<BB * NHD, 192, 0, stream>>>(qb, kb, vb, attn_mask, a);
    gemm512_kernel<2><<<ggrid, 256, 0, stream>>>(a, Wl + 3 * HH * HH, Bp[9], x, nullptr);
    ln_bf16_kernel<<<MM / 8, 256, 0, stream>>>(x, Bp[10], Bp[11], qb);
    gemm512_kernel<1><<<ggrid, 256, 0, stream>>>(qb, Wl + 4 * HH * HH, Bp[13], nullptr, kb);
    gemm512_kernel<2><<<ggrid, 256, 0, stream>>>(kb, Wl + 5 * HH * HH, Bp[15], x, nullptr);
  }
  heads_kernel<<<BB * TT, 32, 0, stream>>>(x, F[113], F[114], F[115], F[116],
                                           (float*)d_out);
}
